// CausalSelfAttention_56659208569187
// MI455X (gfx1250) — matrix operand order repaired and checked by reading, not compiled, not run on hardware
//
#include <hip/hip_runtime.h>
#include <hip/hip_bf16.h>
#include <stdint.h>

typedef __bf16 bf16;
typedef __attribute__((ext_vector_type(16))) __bf16 v16bf;
typedef __attribute__((ext_vector_type(8)))  __bf16 v8bf;
typedef __attribute__((ext_vector_type(8)))  float  v8f;
typedef __attribute__((ext_vector_type(4)))  unsigned int v4u;
typedef __attribute__((ext_vector_type(8)))  int v8i;
typedef __attribute__((ext_vector_type(4)))  int v4i;

#define B_      2
#define T_      2048
#define C_      2048
#define NH_     16
#define NKV_    4
#define HD_     128
#define GATE_CH_ 12
#define WINDOW_ 1024

#define CAT16(lo, hi) __builtin_shufflevector((lo), (hi), 0,1,2,3,4,5,6,7,8,9,10,11,12,13,14,15)

// ---------------------------------------------------------------------------
// TDM: 2-D tensor tile -> LDS (CDNA5 ISA ch.7 "Tensor Data Mover", D# per
// ISA 8.3/8.4). data_size=2B; LDS padding 4 DWORDs every 16 DWORDs so each
// 64B tile row lands on an 80B stride (16B aligned, bank-conflict free).
// ---------------------------------------------------------------------------
__device__ __forceinline__ void tdm_load_2d(uint32_t lds_off, const void* gaddr,
                                            uint32_t tile0, uint32_t tile1,
                                            uint64_t stride0_elems) {
    const uint64_t ga = (uint64_t)(uintptr_t)gaddr;
    v4u g0;
    g0[0] = 1u;                                   // count=1 (valid user D#)
    g0[1] = lds_off;                              // lds byte address
    g0[2] = (uint32_t)ga;                         // global addr [95:64]
    g0[3] = (uint32_t)((ga >> 32) & 0x1FFFFFFu)   // global addr [120:96]
          | (2u << 30);                           // type = 2 ("image")
    v8i g1;
    const uint32_t dw0 = (1u << 16)               // data_size = 2 bytes
                       | (1u << 20)               // pad_enable
                       | (3u << 22)               // pad_interval = 16 DWORDs
                       | (3u << 25);              // pad_amount  = 4 DWORDs
    g1[0] = (int)dw0;
    g1[1] = (int)((tile0 & 0xFFFFu) << 16);                        // tensor_dim0 lo
    g1[2] = (int)(((tile0 >> 16) & 0xFFFFu) | ((tile1 & 0xFFFFu) << 16)); // dim0 hi | dim1 lo
    g1[3] = (int)(((tile1 >> 16) & 0xFFFFu) | ((tile0 & 0xFFFFu) << 16)); // dim1 hi | tile_dim0
    g1[4] = (int)(tile1 & 0xFFFFu);                                // tile_dim1 (tile_dim2=0)
    g1[5] = (int)(uint32_t)(stride0_elems & 0xFFFFFFFFu);          // dim0 stride lo
    g1[6] = (int)(uint32_t)((stride0_elems >> 32) & 0xFFFFu);      // dim0 stride hi
    g1[7] = 0;
    const v4i gz = {0, 0, 0, 0};
#if __clang_major__ >= 23
    const v8i gz8 = {0, 0, 0, 0, 0, 0, 0, 0};
    __builtin_amdgcn_tensor_load_to_lds(g0, g1, gz, gz, gz8, 0);
#else
    __builtin_amdgcn_tensor_load_to_lds(g0, g1, gz, gz, 0);
#endif
}

// ---------------------------------------------------------------------------
// fp32 -> bf16 conversion
// ---------------------------------------------------------------------------
__global__ void cvt_f32_bf16(const float* __restrict__ in, bf16* __restrict__ out, int n) {
    int i = blockIdx.x * blockDim.x + threadIdx.x;
    if (i < n) out[i] = (bf16)in[i];
}

// ---------------------------------------------------------------------------
// LDS-tiled bf16 GEMM: out[M,N] (f32) = A[M,K] * W[N,K]^T.
// Workgroup (256 thr / 8 waves) computes a 128x128 tile. Per 32-wide K-step,
// wave 0 TDM-loads A(128x32) and W(128x32) into double-buffered LDS (80B row
// stride). Each wave computes 32x64: 2 A-frags x 4 B-frags -> 8 WMMAs.
// ---------------------------------------------------------------------------
#define ROWB 80                 // padded LDS row stride in bytes (32 bf16 + 16B)
#define BUFB 10240              // 128 rows * 80B
__launch_bounds__(256)
__global__ void gemm_bf16(const bf16* __restrict__ A, const bf16* __restrict__ W,
                          float* __restrict__ out, int M, int N, int K) {
    __shared__ __align__(16) char smem[4 * BUFB];   // A0 A1 B0 B1

    const int wave = threadIdx.x >> 5;
    const int lane = threadIdx.x & 31;
    const int half = lane >> 4;
    const int lm   = lane & 15;

    const int blockN = blockIdx.x;      // N/128
    const int blockM = blockIdx.y;      // M/128
    const int m0 = (wave & 3) * 32;     // wave's M offset in tile
    const int n0 = (wave >> 2) * 64;    // wave's N offset in tile

    const bf16* Abase = A + (size_t)(blockM * 128) * K;
    const bf16* Wbase = W + (size_t)(blockN * 128) * K;
    const uint32_t lds0 = (uint32_t)(uintptr_t)&smem[0];

    const int nk = K >> 5;

    // prologue: stage k-step 0 into buffer 0
    if (wave == 0) {
        tdm_load_2d(lds0,            Abase, 32u, 128u, (uint64_t)K);
        tdm_load_2d(lds0 + 2 * BUFB, Wbase, 32u, 128u, (uint64_t)K);
        __builtin_amdgcn_s_wait_tensorcnt((short)0);
    }
    __syncthreads();

    v8f acc[2][4];
    #pragma unroll
    for (int ms = 0; ms < 2; ++ms)
        #pragma unroll
        for (int ns = 0; ns < 4; ++ns) acc[ms][ns] = (v8f){};

    for (int ks = 0; ks < nk; ++ks) {
        const int cur = ks & 1;
        if (wave == 0 && ks + 1 < nk) {
            const int nxt = cur ^ 1;
            tdm_load_2d(lds0 + nxt * BUFB,       Abase + (size_t)(ks + 1) * 32, 32u, 128u, (uint64_t)K);
            tdm_load_2d(lds0 + (2 + nxt) * BUFB, Wbase + (size_t)(ks + 1) * 32, 32u, 128u, (uint64_t)K);
        }

        const char* bufA = smem + cur * BUFB;
        const char* bufB = smem + (2 + cur) * BUFB;

        v16bf af[2];
        #pragma unroll
        for (int ms = 0; ms < 2; ++ms) {
            const char* p = bufA + (m0 + ms * 16 + lm) * ROWB + half * 32;
            af[ms] = CAT16(*(const v8bf*)p, *(const v8bf*)(p + 16));
        }
        v16bf bfg[4];
        #pragma unroll
        for (int ns = 0; ns < 4; ++ns) {
            const char* p = bufB + (n0 + ns * 16 + lm) * ROWB + half * 32;
            bfg[ns] = CAT16(*(const v8bf*)p, *(const v8bf*)(p + 16));
        }
        #pragma unroll
        for (int ns = 0; ns < 4; ++ns)
            #pragma unroll
            for (int ms = 0; ms < 2; ++ms)
                acc[ms][ns] = __builtin_amdgcn_wmma_f32_16x16x32_bf16(
                    false, af[ms], false, bfg[ns], (short)0, acc[ms][ns], false, false);

        if (wave == 0) __builtin_amdgcn_s_wait_tensorcnt((short)0);
        __syncthreads();
    }

    #pragma unroll
    for (int ms = 0; ms < 2; ++ms)
        #pragma unroll
        for (int ns = 0; ns < 4; ++ns) {
            float* orow = out + (size_t)(blockM * 128 + m0 + ms * 16 + 8 * half) * N
                              + blockN * 128 + n0 + ns * 16 + lm;
            #pragma unroll
            for (int r = 0; r < 8; ++r) orow[(size_t)r * N] = acc[ms][ns][r];
        }
}

// ---------------------------------------------------------------------------
// RoPE + RMSNorm (*1.2) -> bf16 [B, H, T, HD]
// ---------------------------------------------------------------------------
__launch_bounds__(128)
__global__ void rope_rms(const float* __restrict__ qkf, const float* __restrict__ cs,
                         const float* __restrict__ sn, bf16* __restrict__ outh,
                         int nheads) {
    const int blk = blockIdx.x;
    const int h  = blk % nheads;
    const int bt = blk / nheads;
    const int t  = bt % T_;
    const int b  = bt / T_;
    const int d  = threadIdx.x;

    const float* row = qkf + (size_t)bt * (nheads * HD_) + h * HD_;
    const int i = d & 63;
    const float x1 = row[i];
    const float x2 = row[i + 64];
    const float c  = cs[t * 64 + i];
    const float s  = sn[t * 64 + i];
    const float val = (d < 64) ? (x1 * c + x2 * s) : (-x1 * s + x2 * c);

    __shared__ float red[128];
    red[d] = val * val;
    __syncthreads();
    for (int off = 64; off > 0; off >>= 1) {
        if (d < off) red[d] += red[d + off];
        __syncthreads();
    }
    const float scale = rsqrtf(red[0] * (1.0f / 128.0f) + 1e-6f) * 1.2f;
    outh[(((size_t)(b * nheads + h)) * T_ + t) * HD_ + d] = (bf16)(val * scale);
}

// ---------------------------------------------------------------------------
// v = v + 3*sigmoid(x[:,:12] @ Wgate^T) * ve -> bf16 TRANSPOSED [B, KV, HD, T]
// (transposed so the attention P*V B-fragment reads contiguously along T)
// ---------------------------------------------------------------------------
__launch_bounds__(128)
__global__ void gate_v(const float* __restrict__ vf, const float* __restrict__ x,
                       const float* __restrict__ ve, const float* __restrict__ Wg,
                       bf16* __restrict__ vt) {
    const int blk = blockIdx.x;          // (b*T + t)*NKV + kv
    const int kv  = blk % NKV_;
    const int bt  = blk / NKV_;
    const int t   = bt % T_;
    const int b   = bt / T_;
    const int d   = threadIdx.x;

    const float* xr = x + (size_t)bt * C_;
    float dot = 0.0f;
    #pragma unroll
    for (int cc = 0; cc < GATE_CH_; ++cc) dot += xr[cc] * Wg[kv * GATE_CH_ + cc];
    const float gate = 3.0f / (1.0f + __expf(-dot));

    const size_t vi = (size_t)bt * (NKV_ * HD_) + kv * HD_ + d;
    const float v = vf[vi] + gate * ve[vi];
    vt[((size_t)(b * NKV_ + kv) * HD_ + d) * T_ + t] = (bf16)v;
}

// ---------------------------------------------------------------------------
// Sliding-window causal flash attention, GQA (head h uses kv = h/4).
// One wave per 16-query tile; key blocks of 32; online softmax.
// P restaged via bf16 LDS with 80B padded rows -> aligned ds_load_b128.
// ---------------------------------------------------------------------------
#define PROWB 40   // P row stride in bf16 elements (80 bytes)
__launch_bounds__(128)
__global__ void attn(const bf16* __restrict__ qh, const bf16* __restrict__ kh,
                     const bf16* __restrict__ vt, bf16* __restrict__ y) {
    __shared__ __align__(16) bf16 ldsP[4][16 * PROWB];

    const int wave = threadIdx.x >> 5;
    const int lane = threadIdx.x & 31;
    const int half = lane >> 4;
    const int lm   = lane & 15;

    const int tilesPerHead = T_ / 16;                 // 128
    const int tileIdx = blockIdx.x * 4 + wave;
    if (tileIdx >= B_ * NH_ * tilesPerHead) return;
    const int qt = tileIdx % tilesPerHead;
    const int bh = tileIdx / tilesPerHead;            // b*NH + h
    const int h  = bh % NH_;
    const int b  = bh / NH_;
    const int kv = h / (NH_ / NKV_);
    const int qt0 = qt * 16;

    const bf16* qbase = qh + (((size_t)bh) * T_ + qt0 + lm) * HD_;
    const bf16* kbase = kh + ((size_t)(b * NKV_ + kv)) * T_ * HD_;
    const bf16* vbT   = vt + ((size_t)(b * NKV_ + kv)) * HD_ * T_;

    v16bf qa[4];
    #pragma unroll
    for (int dc = 0; dc < 4; ++dc) {
        const bf16* p = qbase + dc * 32 + half * 16;
        qa[dc] = CAT16(*(const v8bf*)p, *(const v8bf*)(p + 8));
    }

    v8f acc[8];
    #pragma unroll
    for (int i = 0; i < 8; ++i) acc[i] = (v8f){};
    float mrun[8], lrun[8];
    #pragma unroll
    for (int r = 0; r < 8; ++r) { mrun[r] = -1e30f; lrun[r] = 0.0f; }

    int kb_lo = qt0 - WINDOW_; if (kb_lo < 0) kb_lo = 0;
    kb_lo &= ~31;
    const int kb_hi = qt0 + 15;
    const float sc = 0.08838834764831845f;            // 1/sqrt(128)
    bf16* P = ldsP[wave];

    for (int kb = kb_lo; kb <= kb_hi; kb += 32) {
        // ---- S = Q K^T for two 16-key tiles -------------------------------
        v8f s0 = {}, s1 = {};
        #pragma unroll
        for (int dc = 0; dc < 4; ++dc) {
            const bf16* k0p = kbase + (size_t)(kb + lm)      * HD_ + dc * 32 + half * 16;
            const bf16* k1p = kbase + (size_t)(kb + 16 + lm) * HD_ + dc * 32 + half * 16;
            v16bf kb0v = CAT16(*(const v8bf*)k0p, *(const v8bf*)(k0p + 8));
            v16bf kb1v = CAT16(*(const v8bf*)k1p, *(const v8bf*)(k1p + 8));
            s0 = __builtin_amdgcn_wmma_f32_16x16x32_bf16(false, qa[dc], false, kb0v, (short)0, s0, false, false);
            s1 = __builtin_amdgcn_wmma_f32_16x16x32_bf16(false, qa[dc], false, kb1v, (short)0, s1, false, false);
        }

        // ---- scale + sliding-window causal mask ---------------------------
        float tmax[8];
        #pragma unroll
        for (int r = 0; r < 8; ++r) {
            const int t = qt0 + r + 8 * half;
            const int i0 = kb + lm;
            const int i1 = kb + 16 + lm;
            float e0 = s0[r] * sc, e1 = s1[r] * sc;
            e0 = ((i0 <= t) && (t - i0 <= WINDOW_)) ? e0 : -1e30f;
            e1 = ((i1 <= t) && (t - i1 <= WINDOW_)) ? e1 : -1e30f;
            s0[r] = e0; s1[r] = e1;
            tmax[r] = fmaxf(e0, e1);
        }
        #pragma unroll
        for (int off = 1; off < 16; off <<= 1) {
            #pragma unroll
            for (int r = 0; r < 8; ++r) tmax[r] = fmaxf(tmax[r], __shfl_xor(tmax[r], off, 32));
        }

        // ---- online softmax update ----------------------------------------
        float psum[8];
        #pragma unroll
        for (int r = 0; r < 8; ++r) {
            const float newm = fmaxf(mrun[r], tmax[r]);
            const float corr = __expf(mrun[r] - newm);
            const float p0 = __expf(s0[r] - newm);
            const float p1 = __expf(s1[r] - newm);
            s0[r] = p0; s1[r] = p1;
            mrun[r] = newm;
            lrun[r] *= corr;
            psum[r] = p0 + p1;
            #pragma unroll
            for (int dt = 0; dt < 8; ++dt) acc[dt][r] *= corr;
        }
        #pragma unroll
        for (int off = 1; off < 16; off <<= 1) {
            #pragma unroll
            for (int r = 0; r < 8; ++r) psum[r] += __shfl_xor(psum[r], off, 32);
        }
        #pragma unroll
        for (int r = 0; r < 8; ++r) lrun[r] += psum[r];

        // ---- restage P (C-layout f32 regs -> A-layout bf16) via LDS -------
        #pragma unroll
        for (int r = 0; r < 8; ++r) {
            P[(r + 8 * half) * PROWB + lm]      = (bf16)s0[r];
            P[(r + 8 * half) * PROWB + 16 + lm] = (bf16)s1[r];
        }
        asm volatile("s_wait_dscnt 0" ::: "memory");
        const bf16* pp = P + lm * PROWB + half * 16;
        v16bf pa = CAT16(*(const v8bf*)pp, *(const v8bf*)(pp + 8));
        asm volatile("s_wait_dscnt 0" ::: "memory");

        // ---- acc += P V  (V transposed: contiguous along keys) ------------
        #pragma unroll
        for (int dt = 0; dt < 8; ++dt) {
            const bf16* vp = vbT + (size_t)(dt * 16 + lm) * T_ + kb + half * 16;
            v16bf vb = CAT16(*(const v8bf*)vp, *(const v8bf*)(vp + 8));
            acc[dt] = __builtin_amdgcn_wmma_f32_16x16x32_bf16(false, pa, false, vb, (short)0, acc[dt], false, false);
        }
    }

    // ---- epilogue ---------------------------------------------------------
    #pragma unroll
    for (int dt = 0; dt < 8; ++dt) {
        #pragma unroll
        for (int r = 0; r < 8; ++r) {
            const int t = qt0 + r + 8 * half;
            const float o = acc[dt][r] / lrun[r];
            y[((size_t)(b * T_ + t)) * (NH_ * HD_) + h * HD_ + dt * 16 + lm] = (bf16)o;
        }
    }
}

// ---------------------------------------------------------------------------
extern "C" void kernel_launch(void* const* d_in, const int* in_sizes, int n_in,
                              void* d_out, int out_size, void* d_ws, size_t ws_size,
                              hipStream_t stream) {
    (void)in_sizes; (void)n_in; (void)out_size; (void)ws_size;

    const float* x     = (const float*)d_in[0];
    const float* ve    = (const float*)d_in[1];
    const float* cosp  = (const float*)d_in[2];
    const float* sinp  = (const float*)d_in[3];
    const float* Wq    = (const float*)d_in[4];
    const float* Wk    = (const float*)d_in[5];
    const float* Wv    = (const float*)d_in[6];
    const float* Wproj = (const float*)d_in[7];
    const float* Wgate = (const float*)d_in[8];
    float* out = (float*)d_out;

    const int BT = B_ * T_;                       // 4096

    char* ws = (char*)d_ws;
    size_t off = 0;
    auto alloc = [&](size_t bytes) -> void* {
        void* p = ws + off;
        off = (off + bytes + 255) & ~(size_t)255;
        return p;
    };
    bf16* xb  = (bf16*)alloc((size_t)BT * C_ * 2);
    bf16* Wqb = (bf16*)alloc((size_t)NH_  * HD_ * C_ * 2);
    bf16* Wkb = (bf16*)alloc((size_t)NKV_ * HD_ * C_ * 2);
    bf16* Wvb = (bf16*)alloc((size_t)NKV_ * HD_ * C_ * 2);
    bf16* Wpb = (bf16*)alloc((size_t)C_ * C_ * 2);
    float* qf = (float*)alloc((size_t)BT * NH_  * HD_ * 4);
    float* kf = (float*)alloc((size_t)BT * NKV_ * HD_ * 4);
    float* vf = (float*)alloc((size_t)BT * NKV_ * HD_ * 4);
    bf16* qhb = (bf16*)alloc((size_t)BT * NH_  * HD_ * 2);
    bf16* khb = (bf16*)alloc((size_t)BT * NKV_ * HD_ * 2);
    bf16* vtb = (bf16*)alloc((size_t)BT * NKV_ * HD_ * 2);
    bf16* yb  = (bf16*)alloc((size_t)BT * C_ * 2);

    auto cvt = [&](const float* src, bf16* dst, int n) {
        cvt_f32_bf16<<<(n + 255) / 256, 256, 0, stream>>>(src, dst, n);
    };
    cvt(x,     xb,  BT * C_);
    cvt(Wq,    Wqb, NH_  * HD_ * C_);
    cvt(Wk,    Wkb, NKV_ * HD_ * C_);
    cvt(Wv,    Wvb, NKV_ * HD_ * C_);
    cvt(Wproj, Wpb, C_ * C_);

    // QKV projections (TDM-staged, LDS-tiled WMMA GEMMs; 128x128 / workgroup)
    gemm_bf16<<<dim3((NH_ * HD_) / 128, BT / 128), 256, 0, stream>>>(xb, Wqb, qf, BT, NH_ * HD_, C_);
    gemm_bf16<<<dim3((NKV_ * HD_) / 128, BT / 128), 256, 0, stream>>>(xb, Wkb, kf, BT, NKV_ * HD_, C_);
    gemm_bf16<<<dim3((NKV_ * HD_) / 128, BT / 128), 256, 0, stream>>>(xb, Wvb, vf, BT, NKV_ * HD_, C_);

    // RoPE + RMSNorm for q/k; gated value mix (writes V transposed)
    rope_rms<<<BT * NH_,  128, 0, stream>>>(qf, cosp, sinp, qhb, NH_);
    rope_rms<<<BT * NKV_, 128, 0, stream>>>(kf, cosp, sinp, khb, NKV_);
    gate_v<<<BT * NKV_, 128, 0, stream>>>(vf, x, ve, Wgate, vtb);

    // Flash attention (wave per 16-query tile)
    {
        int tiles = B_ * NH_ * (T_ / 16);
        attn<<<(tiles + 3) / 4, 128, 0, stream>>>(qhb, khb, vtb, yb);
    }

    // Output projection
    gemm_bf16<<<dim3(C_ / 128, BT / 128), 256, 0, stream>>>(yb, Wpb, out, BT, C_, C_);
}
